// BnnCIFAR10Model_53369263620401
// MI455X (gfx1250) — 8-bit matrix operand order repaired and checked by reading, compiled, not run on hardware
//
#include <hip/hip_runtime.h>
#include <hip/hip_bf16.h>
#include <stdint.h>


typedef __attribute__((ext_vector_type(8))) int v8i;
typedef __attribute__((ext_vector_type(4))) int v4i;
typedef __attribute__((ext_vector_type(2))) int v2i;

__device__ __forceinline__ signed char sgn8(float v) {
    return v >= 0.0f ? (signed char)1 : (signed char)-1;
}

__global__ void k_sign_nchw_nhwc(const float* __restrict__ in,
                                 signed char* __restrict__ out,
                                 int B, int C, int H, int W, size_t total) {
    size_t idx = (size_t)blockIdx.x * blockDim.x + threadIdx.x;
    if (idx >= total) return;
    int x = (int)(idx % W);
    size_t t = idx / W;
    int y = (int)(t % H);  t /= H;
    int c = (int)(t % C);
    int b = (int)(t / C);
    out[(((size_t)b * H + y) * W + x) * C + c] = sgn8(in[idx]);
}

__global__ void k_sign_convw(const float* __restrict__ w,
                             signed char* __restrict__ out,
                             int Cout, int Kin, int Kpad, size_t total) {
    size_t idx = (size_t)blockIdx.x * blockDim.x + threadIdx.x;
    if (idx >= total) return;
    int k  = (int)(idx % Kpad);
    int co = (int)(idx / Kpad);
    out[idx] = (k < Kin) ? sgn8(w[(size_t)co * Kin + k]) : (signed char)0;
}

__global__ void k_sign_fcw(const float* __restrict__ w,
                           signed char* __restrict__ out,
                           int N, int K, size_t total) {
    size_t idx = (size_t)blockIdx.x * blockDim.x + threadIdx.x;
    if (idx >= total) return;
    int n = (int)(idx / K);
    int k = (int)(idx % K);
    out[idx] = (n < N) ? sgn8(w[(size_t)n * K + k]) : (signed char)0;
}

__global__ void k_im2col(const signed char* __restrict__ act,
                         signed char* __restrict__ col,
                         int B, int H, int W, int C, int Kpad, size_t total) {
    size_t idx = (size_t)blockIdx.x * blockDim.x + threadIdx.x;
    if (idx >= total) return;
    int k = (int)(idx % Kpad);
    size_t row = idx / Kpad;
    int x = (int)(row % W);
    size_t t = row / W;
    int y = (int)(t % H);
    int b = (int)(t / H);
    signed char v = 0;
    int Kin = C * 9;
    if (k < Kin) {
        int cin = k / 9, r = k % 9;
        int iy = y + r / 3 - 1;
        int ix = x + r % 3 - 1;
        if (iy >= 0 && iy < H && ix >= 0 && ix < W)
            v = act[(((size_t)b * H + iy) * W + ix) * C + cin];
    }
    col[idx] = v;
}

template <int NT>
__global__ void k_bgemm_wmma(const signed char* __restrict__ A,
                             const signed char* __restrict__ Bm,
                             const float* __restrict__ bias,
                             float* __restrict__ C,
                             int Kpad, int ldC, int Nvalid, int ntw) {
    constexpr int KC = 512;
    __shared__ signed char sB[NT * 16 * KC];

    const int tnw    = blockIdx.x % ntw;
    const int tm     = (blockIdx.x / ntw) * 8 + (threadIdx.x >> 5);
    const int lane   = threadIdx.x & 31;
    const int half   = lane >> 4;
    const int l      = lane & 15;
    const int nBase  = tnw * (NT * 16);

    const signed char* Arow = A + (size_t)(tm * 16 + l) * Kpad;

    v8i acc[NT];
#pragma unroll
    for (int j = 0; j < NT; ++j) acc[j] = (v8i){0, 0, 0, 0, 0, 0, 0, 0};

    for (int kc0 = 0; kc0 < Kpad; kc0 += KC) {
        const int kcLen = (Kpad - kc0 < KC) ? (Kpad - kc0) : KC;

        {
            const int vecs = (NT * 16 * kcLen) >> 4;
            for (int v = threadIdx.x; v < vecs; v += blockDim.x) {
                const int byteOff = v << 4;
                const int r = byteOff / kcLen;
                const int c = byteOff % kcLen;
                const signed char* g = Bm + (size_t)(nBase + r) * Kpad + kc0 + c;
                unsigned ldsOff = (unsigned)(uintptr_t)(&sB[r * KC + c]);
                asm volatile("global_load_async_to_lds_b128 %0, %1, off"
                             :: "v"(ldsOff), "v"(g) : "memory");
            }
            asm volatile("s_wait_asynccnt 0x0" ::: "memory");
        }
        __syncthreads();

        const int ksteps = kcLen >> 6;
        for (int ks = 0; ks < ksteps; ++ks) {
            const int k0 = kc0 + (ks << 6);
            const int kl = ks << 6;
            __builtin_prefetch(Arow + k0 + 64, 0, 1);
            const v4i* Ap = (const v4i*)(Arow + k0);
            v4i a0 = Ap[half];
            v4i a1 = Ap[2 + half];
            v8i a;
            a[0] = a0[0]; a[1] = a0[1];
            a[2] = a0[2]; a[3] = a0[3];
            a[4] = a1[0]; a[5] = a1[1];
            a[6] = a1[2]; a[7] = a1[3];
#pragma unroll
            for (int j = 0; j < NT; ++j) {
                const v4i* Bp = (const v4i*)(&sB[(j * 16 + l) * KC + kl]);
                v4i b0 = Bp[half];
                v4i b1 = Bp[2 + half];
                v8i b;
                b[0] = b0[0]; b[1] = b0[1]; b[2] = b0[2]; b[3] = b0[3];
                b[4] = b1[0]; b[5] = b1[1]; b[6] = b1[2]; b[7] = b1[3];
                acc[j] = __builtin_amdgcn_wmma_i32_16x16x64_iu8(
                    true, a, true, b, acc[j], false, false);
            }
        }
        __syncthreads();
    }

#pragma unroll
    for (int j = 0; j < NT; ++j) {
        int n = nBase + j * 16 + l;
        if (n < Nvalid) {
            float bs = bias[n];
            float* Crow = C + (size_t)(tm * 16) * ldC + n;
#pragma unroll
            for (int r = 0; r < 8; ++r)
                Crow[(size_t)(r + half * 8) * ldC] = (float)acc[j][r] + bs;
        }
    }
}

__global__ void k_relu_pool_sign(const float* __restrict__ in,
                                 signed char* __restrict__ out,
                                 int B, int H, int W, int C, size_t total) {
    size_t idx = (size_t)blockIdx.x * blockDim.x + threadIdx.x;
    if (idx >= total) return;
    int c = (int)(idx % C);
    size_t t = idx / C;
    int Wo = W >> 1, Ho = H >> 1;
    int xo = (int)(t % Wo); t /= Wo;
    int yo = (int)(t % Ho);
    int b  = (int)(t / Ho);
    size_t r00 = ((size_t)b * H + 2 * yo) * W + 2 * xo;
    float v0 = in[r00 * C + c];
    float v1 = in[(r00 + 1) * C + c];
    float v2 = in[(r00 + W) * C + c];
    float v3 = in[(r00 + W + 1) * C + c];
    float m = fmaxf(fmaxf(v0, v1), fmaxf(v2, v3));
    float rp = fmaxf(m, 0.0f);
    out[idx] = (rp >= 0.0f) ? (signed char)1 : (signed char)-1;
}

__global__ void k_flatten_nchw(const signed char* __restrict__ in,
                               signed char* __restrict__ out, size_t total) {
    size_t idx = (size_t)blockIdx.x * blockDim.x + threadIdx.x;
    if (idx >= total) return;
    size_t b = idx >> 11;
    int k = (int)(idx & 2047);
    int c = k >> 4;
    int r = k & 15;
    int y = r >> 2;
    int x = r & 3;
    out[idx] = in[(((b * 4 + y) * 4 + x) << 7) + c];
}

__global__ void k_relu_sign(const float* __restrict__ in,
                            signed char* __restrict__ out, size_t total) {
    size_t idx = (size_t)blockIdx.x * blockDim.x + threadIdx.x;
    if (idx >= total) return;
    float v = fmaxf(in[idx], 0.0f);
    out[idx] = (v >= 0.0f) ? (signed char)1 : (signed char)-1;
}

static inline int nblk(size_t total, int tpb) {
    return (int)((total + (size_t)tpb - 1) / (size_t)tpb);
}

extern "C" void kernel_launch(void* const* d_in, const int* in_sizes, int n_in,
                              void* d_out, int out_size, void* d_ws, size_t ws_size,
                              hipStream_t stream) {
    const float* x   = (const float*)d_in[0];
    const float* w1  = (const float*)d_in[1];
    const float* b1  = (const float*)d_in[2];
    const float* w2  = (const float*)d_in[3];
    const float* b2  = (const float*)d_in[4];
    const float* w3  = (const float*)d_in[5];
    const float* b3  = (const float*)d_in[6];
    const float* fw1 = (const float*)d_in[7];
    const float* fb1 = (const float*)d_in[8];
    const float* fw2 = (const float*)d_in[9];
    const float* fb2 = (const float*)d_in[10];
    float* out = (float*)d_out;

    const int B = in_sizes[0] / (3 * 32 * 32);

    char* ws = (char*)d_ws;
    size_t off = 0;
    auto carve = [&](size_t bytes) -> char* {
        char* p = ws + off;
        off += (bytes + 255) & ~(size_t)255;
        return p;
    };
    signed char* act0  = (signed char*)carve((size_t)B * 32 * 32 * 3);
    signed char* col   = (signed char*)carve((size_t)B * 256 * 320);
    float*       gout  = (float*)carve((size_t)B * 1024 * 32 * 4);
    signed char* act1  = (signed char*)carve((size_t)B * 16 * 16 * 32);
    signed char* act2  = (signed char*)carve((size_t)B * 8 * 8 * 64);
    signed char* act3p = (signed char*)carve((size_t)B * 4 * 4 * 128);
    signed char* act3  = (signed char*)carve((size_t)B * 2048);
    signed char* act4  = (signed char*)carve((size_t)B * 512);
    signed char* wb1   = (signed char*)carve(32 * 64);
    signed char* wb2   = (signed char*)carve(64 * 320);
    signed char* wb3   = (signed char*)carve((size_t)128 * 576);
    signed char* fwb1  = (signed char*)carve((size_t)512 * 2048);
    signed char* fwb2  = (signed char*)carve((size_t)16 * 512);

    const int TPB = 256;

    { size_t t = 32 * 64;
      k_sign_convw<<<nblk(t, TPB), TPB, 0, stream>>>(w1, wb1, 32, 27, 64, t); }
    { size_t t = 64 * 320;
      k_sign_convw<<<nblk(t, TPB), TPB, 0, stream>>>(w2, wb2, 64, 288, 320, t); }
    { size_t t = (size_t)128 * 576;
      k_sign_convw<<<nblk(t, TPB), TPB, 0, stream>>>(w3, wb3, 128, 576, 576, t); }
    { size_t t = (size_t)512 * 2048;
      k_sign_fcw<<<nblk(t, TPB), TPB, 0, stream>>>(fw1, fwb1, 512, 2048, t); }
    { size_t t = (size_t)16 * 512;
      k_sign_fcw<<<nblk(t, TPB), TPB, 0, stream>>>(fw2, fwb2, 10, 512, t); }

    { size_t t = (size_t)B * 3 * 32 * 32;
      k_sign_nchw_nhwc<<<nblk(t, TPB), TPB, 0, stream>>>(x, act0, B, 3, 32, 32, t); }
    { size_t t = (size_t)B * 1024 * 64;
      k_im2col<<<nblk(t, TPB), TPB, 0, stream>>>(act0, col, B, 32, 32, 3, 64, t); }
    { int M = B * 1024, ntw = 1;
      k_bgemm_wmma<2><<<(M / 128) * ntw, TPB, 0, stream>>>(
          col, wb1, b1, gout, 64, 32, 32, ntw); }
    { size_t t = (size_t)B * 16 * 16 * 32;
      k_relu_pool_sign<<<nblk(t, TPB), TPB, 0, stream>>>(gout, act1, B, 32, 32, 32, t); }

    { size_t t = (size_t)B * 256 * 320;
      k_im2col<<<nblk(t, TPB), TPB, 0, stream>>>(act1, col, B, 16, 16, 32, 320, t); }
    { int M = B * 256, ntw = 1;
      k_bgemm_wmma<4><<<(M / 128) * ntw, TPB, 0, stream>>>(
          col, wb2, b2, gout, 320, 64, 64, ntw); }
    { size_t t = (size_t)B * 8 * 8 * 64;
      k_relu_pool_sign<<<nblk(t, TPB), TPB, 0, stream>>>(gout, act2, B, 16, 16, 64, t); }

    { size_t t = (size_t)B * 64 * 576;
      k_im2col<<<nblk(t, TPB), TPB, 0, stream>>>(act2, col, B, 8, 8, 64, 576, t); }
    { int M = B * 64, ntw = 2;
      k_bgemm_wmma<4><<<(M / 128) * ntw, TPB, 0, stream>>>(
          col, wb3, b3, gout, 576, 128, 128, ntw); }
    { size_t t = (size_t)B * 4 * 4 * 128;
      k_relu_pool_sign<<<nblk(t, TPB), TPB, 0, stream>>>(gout, act3p, B, 8, 8, 128, t); }

    { size_t t = (size_t)B * 2048;
      k_flatten_nchw<<<nblk(t, TPB), TPB, 0, stream>>>(act3p, act3, t); }

    { int M = B, ntw = 8;
      k_bgemm_wmma<4><<<(M / 128) * ntw, TPB, 0, stream>>>(
          act3, fwb1, fb1, gout, 2048, 512, 512, ntw); }
    { size_t t = (size_t)B * 512;
      k_relu_sign<<<nblk(t, TPB), TPB, 0, stream>>>(gout, act4, t); }

    { int M = B, ntw = 1;
      k_bgemm_wmma<1><<<(M / 128) * ntw, TPB, 0, stream>>>(
          act4, fwb2, fb2, out, 512, 10, 10, ntw); }

    (void)n_in; (void)out_size; (void)ws_size;
}
